// VectorQuantizer_59347858096405
// MI455X (gfx1250) — matrix operand order repaired and checked by reading, compiled, not run on hardware
//
#include <hip/hip_runtime.h>

// ---------------------------------------------------------------------------
// VQ (vector quantizer) for MI455X / gfx1250, wave32 + WMMA bf16x3.
//   score[n][k] = ||c_k||^2 - 2 x_n . c_k        (||x||^2 constant per row)
//   out[n]      = codebook[argmin_k score[n][k]]
// GEMM as 3x V_WMMA_F32_16X16X32_BF16 with (xhi,xlo)x(chi,clo) split; cnorm
// folded into the C-accumulator init. Codebook chunks stream into ping-pong
// LDS buffers via GLOBAL_LOAD_ASYNC_TO_LDS (ASYNCcnt) when available.
// ---------------------------------------------------------------------------

typedef __attribute__((ext_vector_type(16))) __bf16 bf16x16;
typedef __attribute__((ext_vector_type(8)))  float  v8f;
typedef __attribute__((ext_vector_type(4)))  float  f32x4;
typedef __attribute__((ext_vector_type(4)))  int    v4i;

typedef __attribute__((address_space(1))) v4i* gv4i_ptr;  // global
typedef __attribute__((address_space(3))) v4i* lv4i_ptr;  // LDS

#define EMBED   256
#define NCODES  1024
#define NROWS   32768          // 8*4*32*32
#define KC      32             // codes per LDS chunk
#define NCHUNK  (NCODES / KC)  // 32

#ifndef VQ_USE_ASYNC
#define VQ_USE_ASYNC 1
#endif
#if VQ_USE_ASYNC && __has_builtin(__builtin_amdgcn_global_load_async_to_lds_b128)
#define HAVE_ASYNC_LDS 1
#else
#define HAVE_ASYNC_LDS 0
#endif

__device__ __forceinline__ void wait_asynccnt0() {
#if __has_builtin(__builtin_amdgcn_s_wait_asynccnt)
    __builtin_amdgcn_s_wait_asynccnt(0);
#else
    asm volatile("s_wait_asynccnt 0" ::: "memory");
#endif
}

// ---------------------------------------------------------------------------
// Prologue: split -2*codebook into bf16 hi/lo, compute ||c||^2 per code.
// ---------------------------------------------------------------------------
__global__ __launch_bounds__(256) void vq_prep_kernel(
    const float* __restrict__ cb,
    __bf16* __restrict__ chi,
    __bf16* __restrict__ clo,
    float*  __restrict__ cnorm)
{
    const int k = blockIdx.x;
    const int d = threadIdx.x;
    const int i = k * EMBED + d;

    float c = cb[i];
    float t = -2.0f * c;
    __bf16 h = (__bf16)t;
    chi[i] = h;
    clo[i] = (__bf16)(t - (float)h);

    float sq = c * c;
    #pragma unroll
    for (int m = 1; m < 32; m <<= 1) sq += __shfl_xor(sq, m, 32);
    __shared__ float red[8];
    if ((threadIdx.x & 31) == 0) red[threadIdx.x >> 5] = sq;
    __syncthreads();
    if (threadIdx.x == 0) {
        float s = 0.f;
        #pragma unroll
        for (int w = 0; w < 8; ++w) s += red[w];
        cnorm[k] = s;
    }
}

// ---------------------------------------------------------------------------
// Main kernel: 8 waves/block, each wave owns one 16-row m-tile and sweeps all
// 1024 codes. Codebook hi/lo double-buffered through LDS (async when possible).
// ---------------------------------------------------------------------------
__global__ __launch_bounds__(256) void vq_main_kernel(
    const float*  __restrict__ x,
    const float*  __restrict__ cb,
    const __bf16* __restrict__ chi,
    const __bf16* __restrict__ clo,
    const float*  __restrict__ cnorm,
    float* __restrict__ out)
{
    // ping-pong codebook buffers: 2 * (16KB hi + 16KB lo) = 64KB total
    __shared__ __attribute__((aligned(32))) __bf16 sBhi[2][KC][EMBED];
    __shared__ __attribute__((aligned(32))) __bf16 sBlo[2][KC][EMBED];

    const int tid     = threadIdx.x;
    const int lane    = tid & 31;
    const int wave    = tid >> 5;
    const int m_base  = (blockIdx.x * 8 + wave) * 16;
    const int col     = lane & 15;
    const bool hihalf = lane >= 16;

    // ---- A fragments: load 16 x rows once (streaming/NT), bf16 hi/lo split.
    // 16-bit A 16x32 layout (contiguous, matches B): lanes 0-15 row=lane, elems
    // 0..15 = K 0..15; lanes 16-31 same row, K 16..31.
    bf16x16 Ahi[8], Alo[8];
    {
        const float* xrow = x + (size_t)(m_base + col) * EMBED;
        const int aoff = hihalf ? 16 : 0;
        #pragma unroll
        for (int s = 0; s < 8; ++s) {
            f32x4 f0 = __builtin_nontemporal_load((const f32x4*)(xrow + 32 * s + aoff));
            f32x4 f1 = __builtin_nontemporal_load((const f32x4*)(xrow + 32 * s + aoff + 4));
            f32x4 f2 = __builtin_nontemporal_load((const f32x4*)(xrow + 32 * s + aoff + 8));
            f32x4 f3 = __builtin_nontemporal_load((const f32x4*)(xrow + 32 * s + aoff + 12));
            float v[16] = {f0.x, f0.y, f0.z, f0.w, f1.x, f1.y, f1.z, f1.w,
                           f2.x, f2.y, f2.z, f2.w, f3.x, f3.y, f3.z, f3.w};
            #pragma unroll
            for (int e = 0; e < 16; ++e) {
                __bf16 h = (__bf16)v[e];
                Ahi[s][e] = h;
                Alo[s][e] = (__bf16)(v[e] - (float)h);
            }
        }
    }

    float minv[8];
    int   mini[8];
    #pragma unroll
    for (int r = 0; r < 8; ++r) { minv[r] = __builtin_inff(); mini[r] = 0; }

    const int boff = hihalf ? 16 : 0;  // B 32x16: lanes 0-15 see K 0..15, 16-31 see 16..31

#if HAVE_ASYNC_LDS
    // ---- async copy chunk 0 into buffer 0 ----
    {
        const v4i* ghi = (const v4i*)chi;
        const v4i* glo = (const v4i*)clo;
        v4i* bhi = (v4i*)&sBhi[0][0][0];
        v4i* blo = (v4i*)&sBlo[0][0][0];
        #pragma unroll
        for (int j = 0; j < 4; ++j) {
            __builtin_amdgcn_global_load_async_to_lds_b128(
                (gv4i_ptr)(ghi + tid + 256 * j), (lv4i_ptr)(bhi + tid + 256 * j), 0, 0);
            __builtin_amdgcn_global_load_async_to_lds_b128(
                (gv4i_ptr)(glo + tid + 256 * j), (lv4i_ptr)(blo + tid + 256 * j), 0, 0);
        }
    }
    wait_asynccnt0();
    __syncthreads();
#else
    // ---- fallback: register staging, double-buffered ----
    uint4 rh[4], rl[4];
    {
        const uint4* ghi = (const uint4*)chi;
        const uint4* glo = (const uint4*)clo;
        #pragma unroll
        for (int j = 0; j < 4; ++j) {
            rh[j] = ghi[tid + 256 * j];
            rl[j] = glo[tid + 256 * j];
        }
    }
#endif

    for (int kc = 0; kc < NCHUNK; ++kc) {
        const int p = kc & 1;

#if HAVE_ASYNC_LDS
        // kick off next chunk into the other buffer while we compute this one
        if (kc + 1 < NCHUNK) {
            const v4i* ghi = (const v4i*)(chi + (size_t)(kc + 1) * KC * EMBED);
            const v4i* glo = (const v4i*)(clo + (size_t)(kc + 1) * KC * EMBED);
            v4i* bhi = (v4i*)&sBhi[p ^ 1][0][0];
            v4i* blo = (v4i*)&sBlo[p ^ 1][0][0];
            #pragma unroll
            for (int j = 0; j < 4; ++j) {
                __builtin_amdgcn_global_load_async_to_lds_b128(
                    (gv4i_ptr)(ghi + tid + 256 * j), (lv4i_ptr)(bhi + tid + 256 * j), 0, 0);
                __builtin_amdgcn_global_load_async_to_lds_b128(
                    (gv4i_ptr)(glo + tid + 256 * j), (lv4i_ptr)(blo + tid + 256 * j), 0, 0);
            }
        }
#else
        // write staged regs into buffer p, then prefetch next chunk into regs
        {
            uint4* bhi = (uint4*)&sBhi[p][0][0];
            uint4* blo = (uint4*)&sBlo[p][0][0];
            #pragma unroll
            for (int j = 0; j < 4; ++j) {
                bhi[tid + 256 * j] = rh[j];
                blo[tid + 256 * j] = rl[j];
            }
        }
        __syncthreads();
        if (kc + 1 < NCHUNK) {
            const uint4* ghi = (const uint4*)(chi + (size_t)(kc + 1) * KC * EMBED);
            const uint4* glo = (const uint4*)(clo + (size_t)(kc + 1) * KC * EMBED);
            #pragma unroll
            for (int j = 0; j < 4; ++j) {
                rh[j] = ghi[tid + 256 * j];
                rl[j] = glo[tid + 256 * j];
            }
        }
#endif

        // ---- compute: 2 k-tiles, each with 2 independent WMMA chains ----
        #pragma unroll
        for (int kt = 0; kt < KC / 16; ++kt) {
            const int crow = kt * 16 + col;
            const float cn = cnorm[kc * KC + crow];  // L2-resident, coalesced
            v8f acc0, acc1;
            #pragma unroll
            for (int e = 0; e < 8; ++e) { acc0[e] = cn; acc1[e] = 0.0f; }

            #pragma unroll
            for (int s = 0; s < 8; ++s) {
                bf16x16 bh = *(const bf16x16*)&sBhi[p][crow][32 * s + boff];
                bf16x16 bl = *(const bf16x16*)&sBlo[p][crow][32 * s + boff];
                if (s & 1) {
                    acc1 = __builtin_amdgcn_wmma_f32_16x16x32_bf16(
                        false, Ahi[s], false, bh, (short)0, acc1, false, false);
                    acc1 = __builtin_amdgcn_wmma_f32_16x16x32_bf16(
                        false, Ahi[s], false, bl, (short)0, acc1, false, false);
                    acc1 = __builtin_amdgcn_wmma_f32_16x16x32_bf16(
                        false, Alo[s], false, bh, (short)0, acc1, false, false);
                } else {
                    acc0 = __builtin_amdgcn_wmma_f32_16x16x32_bf16(
                        false, Ahi[s], false, bh, (short)0, acc0, false, false);
                    acc0 = __builtin_amdgcn_wmma_f32_16x16x32_bf16(
                        false, Ahi[s], false, bl, (short)0, acc0, false, false);
                    acc0 = __builtin_amdgcn_wmma_f32_16x16x32_bf16(
                        false, Alo[s], false, bh, (short)0, acc0, false, false);
                }
            }

            const int code = kc * KC + kt * 16 + col;
            #pragma unroll
            for (int r = 0; r < 8; ++r) {
                float sc = acc0[r] + acc1[r];
                if (sc < minv[r]) { minv[r] = sc; mini[r] = code; }
            }
        }

#if HAVE_ASYNC_LDS
        wait_asynccnt0();   // next chunk landed in LDS
        __syncthreads();
#else
        __syncthreads();
#endif
    }

    // ---- argmin across the 16 lanes sharing each row (C layout halves) ----
    #pragma unroll
    for (int r = 0; r < 8; ++r) {
        #pragma unroll
        for (int m = 1; m < 16; m <<= 1) {
            float ov = __shfl_xor(minv[r], m, 32);
            int   oi = __shfl_xor(mini[r], m, 32);
            if (ov < minv[r] || (ov == minv[r] && oi < mini[r])) {
                minv[r] = ov; mini[r] = oi;
            }
        }
    }
    // lane 0 holds rows m_base+0..7 (reg r), lane 16 rows m_base+8..15.

    // ---- gather winning codebook rows (f32) to output, NT stores ----
    #pragma unroll
    for (int rl = 0; rl < 16; ++rl) {
        int idx = __shfl(mini[rl & 7], (rl < 8) ? 0 : 16, 32);
        const f32x4* src = (const f32x4*)(cb + (size_t)idx * EMBED);
        f32x4* dst = (f32x4*)(out + (size_t)(m_base + rl) * EMBED);
        __builtin_nontemporal_store(src[lane],      dst + lane);
        __builtin_nontemporal_store(src[lane + 32], dst + lane + 32);
    }
}

// ---------------------------------------------------------------------------
extern "C" void kernel_launch(void* const* d_in, const int* in_sizes, int n_in,
                              void* d_out, int out_size, void* d_ws, size_t ws_size,
                              hipStream_t stream)
{
    const float* x  = (const float*)d_in[0];   // [32768, 256] f32
    const float* cb = (const float*)d_in[1];   // [1024, 256]  f32

    // workspace: chi (512KB) | clo (512KB) | cnorm (4KB)
    __bf16* chi   = (__bf16*)d_ws;
    __bf16* clo   = (__bf16*)((char*)d_ws + (size_t)NCODES * EMBED * 2);
    float*  cnorm = (float*)((char*)d_ws + (size_t)NCODES * EMBED * 4);
    float*  out   = (float*)d_out;

    vq_prep_kernel<<<NCODES, 256, 0, stream>>>(cb, chi, clo, cnorm);
    vq_main_kernel<<<NROWS / (16 * 8), 256, 0, stream>>>(x, cb, chi, clo, cnorm, out);
}
